// MaskedMSA_37915971289109
// MI455X (gfx1250) — compile-verified, matrix operand order checked by reading, not run on hardware
//
#include <hip/hip_runtime.h>

// ---------------------------------------------------------------------------
// MaskedMSA for MI455X (gfx1250): bf16 WMMA (f32 accum) + Tensor Data Mover
//   0) f32 -> bf16 conversion pass for feats / Wqkv / Wp   (memory bound)
//   1) qkv = mask ? x @ Wqkv^T : 0      (WMMA GEMM, TDM-staged, bf16 out)
//   2) flash-attention per (b,h): softmax(q k^T / 8) v     (WMMA + TDM)
//   3) out = mask ? o @ Wp^T + bp : 0   (WMMA GEMM, TDM-staged, f32 out)
// Shapes: B=8 T=1024 C=768 H=12 D=64, N=8+8192 rows.
// ---------------------------------------------------------------------------

typedef __attribute__((ext_vector_type(16))) __bf16    v16bf;
typedef __attribute__((ext_vector_type(8)))  float     v8f;
typedef __attribute__((ext_vector_type(4)))  __bf16    bf4;
typedef __attribute__((ext_vector_type(4)))  unsigned  v4u;
typedef __attribute__((ext_vector_type(8)))  int       v8i;
typedef __attribute__((ext_vector_type(4)))  int       v4i;

#if __has_builtin(__builtin_amdgcn_tensor_load_to_lds) && \
    __has_builtin(__builtin_amdgcn_s_wait_tensorcnt)
#define HAVE_TDM 1
#else
#define HAVE_TDM 0
#endif

__device__ __forceinline__ v8f wmma_bf16(v16bf a, v16bf b, v8f c) {
  // (neg_a, A, neg_b, B, c_mod, C, reuse_a, reuse_b)
  return __builtin_amdgcn_wmma_f32_16x16x32_bf16(false, a, false, b,
                                                 (short)0, c, false, false);
}

#if HAVE_TDM
// Issue one 2-D TDM tile load: global (row-major, `stride` bf16 elems/row,
// tile w x h bf16 elements) -> LDS at byte offset lds_off. D# packed per
// CDNA5 ISA: group0 = {count|flags, lds_addr,
// global_addr lo, global_addr hi | type=2}; group1 = {data_size=2B, dims,
// tile dims, dim0 stride}. Groups 2/3 zero (2-D tensor). Tracked: TENSORcnt.
__device__ __forceinline__ void tdm_load_2d(unsigned lds_off, const void* gptr,
                                            unsigned w, unsigned h,
                                            unsigned stride) {
  unsigned long long ga = (unsigned long long)(size_t)gptr;
  v4u g0;
  g0[0] = 1u;                                                // count=1
  g0[1] = lds_off;                                           // lds_addr
  g0[2] = (unsigned)ga;                                      // gaddr[31:0]
  g0[3] = (unsigned)((ga >> 32) & 0x01ffffffu) | (2u << 30); // gaddr[56:32]|type
  v8i g1;
  g1[0] = (int)(1u << 16);                        // data_size code 1 = 2 bytes
  g1[1] = (int)((w & 0xffffu) << 16);             // tensor_dim0[15:0]
  g1[2] = (int)((w >> 16) | ((h & 0xffffu) << 16));  // dim0 hi | dim1 lo
  g1[3] = (int)((h >> 16) | (w << 16));           // dim1 hi | tile_dim0 = w
  g1[4] = (int)(h & 0xffffu);                     // tile_dim1 = h, tile_dim2=0
  g1[5] = (int)stride;                            // tensor_dim0_stride[31:0]
  g1[6] = 0;                                      // stride hi | dim1_stride lo
  g1[7] = 0;
  v4i z4 = {0, 0, 0, 0};
#if defined(__clang_major__) && __clang_major__ >= 23
  v8i z8 = {0, 0, 0, 0, 0, 0, 0, 0};
  __builtin_amdgcn_tensor_load_to_lds(g0, g1, z4, z4, z8, 0);
#else
  __builtin_amdgcn_tensor_load_to_lds(g0, g1, z4, z4, 0);
#endif
}
#endif

// 16x32 WMMA fragment (A layout; B identical with "row" = output column) from
// an LDS tile, leading dim `ld` bf16 elems. row = (lane&15)+off,
// hi8 = (lane>>4)*8. VGPR r holds K pair k = kbase + hi8 + (r&3)*2 + (r>=4)*16.
__device__ __forceinline__ v16bf frag_ld(const __bf16* tile, int row, int ld,
                                         int kbase, int hi8) {
  union { v16bf v; unsigned u[8]; } f;
  const __bf16* p = tile + row * ld + kbase + hi8;
#pragma unroll
  for (int r = 0; r < 8; ++r) {
    int k = ((r & 3) << 1) + ((r >> 2) << 4);
    f.u[r] = *(const unsigned*)(p + k);
  }
  return f.v;
}

// Same fragment but gathering along the leading dimension (for a row-major
// K x N tile used as B with N = `col`): element pair r covers k, k+1 rows.
__device__ __forceinline__ v16bf frag_ld_col(const __bf16* tile, int col, int ld,
                                             int kbase, int hi8) {
  union { v16bf v; __bf16 e[16]; } f;
  const __bf16* p = tile + col + (kbase + hi8) * ld;
#pragma unroll
  for (int r = 0; r < 8; ++r) {
    int k = ((r & 3) << 1) + ((r >> 2) << 4);
    f.e[2 * r]     = p[k * ld];
    f.e[2 * r + 1] = p[(k + 1) * ld];
  }
  return f.v;
}

// ---------------------------------------------------------------------------
// Kernel 0: elementwise f32 -> bf16 (vectorized x4)
// ---------------------------------------------------------------------------
__global__ __launch_bounds__(256)
void cvt_bf16_kernel(const float* __restrict__ src, __bf16* __restrict__ dst,
                     int n4) {
  int i = blockIdx.x * 256 + threadIdx.x;
  if (i < n4) {
    float4 f = *(const float4*)(src + (size_t)i * 4);
    bf4 h; h[0] = (__bf16)f.x; h[1] = (__bf16)f.y;
           h[2] = (__bf16)f.z; h[3] = (__bf16)f.w;
    *(bf4*)&dst[(size_t)i * 4] = h;
  }
}

// ---------------------------------------------------------------------------
// Kernels 1/3: C[m,n] = sum_k A[m,k]*W[n,k], 128x128 block tile, 256 thr =
// 8 waves, each wave 32x64 (2x4 WMMA frags). Double-buffered TDM staging.
// ---------------------------------------------------------------------------
#define GEMM_BODY(EPILOGUE)                                                    \
  __shared__ __bf16 As[2][128 * 32];                                           \
  __shared__ __bf16 Bs[2][128 * 32];                                           \
  const int tid = threadIdx.x, wave = tid >> 5, lane = tid & 31;               \
  const int wm = wave >> 1, wn = wave & 1;                                     \
  const int hi8 = (lane >> 4) << 3, lo = lane & 15;                            \
  const int m0 = blockIdx.x * 128, n0 = blockIdx.y * 128;                      \
  const int nsteps = K >> 5;                                                   \
  v8f acc[2][4] = {};                                                          \
  auto stage = [&](int s) {                                                    \
    const __bf16* ag = X + (size_t)m0 * K + s * 32;                            \
    const __bf16* bg = W + (size_t)n0 * K + s * 32;                            \
    STAGE_PAIR(ag, bg, s)                                                      \
  };                                                                           \
  stage(0);                                                                    \
  TDM_WAIT0                                                                    \
  __syncthreads();                                                             \
  for (int s = 0; s < nsteps; ++s) {                                           \
    if (s + 1 < nsteps) stage(s + 1);                                          \
    TDM_WAIT_STEP                                                              \
    __syncthreads();                                                           \
    const __bf16* Ab = As[s & 1];                                              \
    const __bf16* Bb = Bs[s & 1];                                              \
    v16bf a[2], bb[4];                                                         \
    _Pragma("unroll")                                                          \
    for (int i = 0; i < 2; ++i)                                                \
      a[i] = frag_ld(Ab, wm * 32 + i * 16 + lo, 32, 0, hi8);                   \
    _Pragma("unroll")                                                          \
    for (int j = 0; j < 4; ++j)                                                \
      bb[j] = frag_ld(Bb, wn * 64 + j * 16 + lo, 32, 0, hi8);                  \
    _Pragma("unroll")                                                          \
    for (int i = 0; i < 2; ++i)                                                \
      _Pragma("unroll")                                                        \
      for (int j = 0; j < 4; ++j) acc[i][j] = wmma_bf16(a[i], bb[j], acc[i][j]);\
    __syncthreads();                                                           \
  }                                                                            \
  _Pragma("unroll")                                                            \
  for (int i = 0; i < 2; ++i)                                                  \
    _Pragma("unroll")                                                          \
    for (int j = 0; j < 4; ++j) {                                              \
      int gcol = n0 + wn * 64 + j * 16 + lo;                                   \
      _Pragma("unroll")                                                        \
      for (int r = 0; r < 8; ++r) {                                            \
        int grow = m0 + wm * 32 + i * 16 + hi8 + r;                            \
        EPILOGUE                                                               \
      }                                                                        \
    }

#if HAVE_TDM
#define STAGE_PAIR(ag, bg, s)                                                  \
  if (tid < 32) {                                                              \
    tdm_load_2d((unsigned)(size_t)&As[(s) & 1][0], ag, 32, 128, (unsigned)K);  \
    tdm_load_2d((unsigned)(size_t)&Bs[(s) & 1][0], bg, 32, 128, (unsigned)K);  \
  }
#define TDM_WAIT0 if (tid < 32) __builtin_amdgcn_s_wait_tensorcnt(0);
#define TDM_WAIT_STEP                                                          \
  if (tid < 32) {                                                              \
    if (s + 1 < nsteps) __builtin_amdgcn_s_wait_tensorcnt(2);                  \
    else                __builtin_amdgcn_s_wait_tensorcnt(0);                  \
  }
#else
#define STAGE_PAIR(ag, bg, s)                                                  \
  for (int t = tid; t < 1024; t += 256) {                                      \
    int row = t >> 3, g = t & 7;                                               \
    *(bf4*)&As[(s) & 1][row * 32 + g * 4] =                                    \
        *(const bf4*)&ag[(size_t)row * K + g * 4];                             \
    *(bf4*)&Bs[(s) & 1][row * 32 + g * 4] =                                    \
        *(const bf4*)&bg[(size_t)row * K + g * 4];                             \
  }
#define TDM_WAIT0
#define TDM_WAIT_STEP
#endif

__global__ __launch_bounds__(256)
void qkv_gemm_kernel(const __bf16* __restrict__ X, const __bf16* __restrict__ W,
                     const unsigned char* __restrict__ mask,
                     __bf16* __restrict__ Y, int K, int N) {
  GEMM_BODY({
    float v = mask[grow] ? acc[i][j][r] : 0.0f;
    Y[(size_t)grow * N + gcol] = (__bf16)v;
  })
}

__global__ __launch_bounds__(256)
void proj_gemm_kernel(const __bf16* __restrict__ X, const __bf16* __restrict__ W,
                      const float* __restrict__ bias,
                      const unsigned char* __restrict__ mask,
                      float* __restrict__ out, int K, int N) {
  GEMM_BODY({
    out[(size_t)grow * N + gcol] =
        mask[grow] ? (acc[i][j][r] + bias[gcol]) : 0.0f;
  })
}

// ---------------------------------------------------------------------------
// Kernel 2: flash attention. Grid (B*H, T/64), 128 threads = 4 waves, each
// wave owns 16 query rows. K/V tiles double-buffered via TDM; V used row-major
// (k x d) with strided B-fragment gathers for the P.V WMMA.
// ---------------------------------------------------------------------------
__global__ __launch_bounds__(128)
void attn_kernel(const __bf16* __restrict__ qkv, __bf16* __restrict__ O) {
  const int T = 1024, C3 = 2304, Hn = 12, Cc = 768;
  const int b = blockIdx.x / Hn, h = blockIdx.x % Hn;
  const int qt = blockIdx.y;
  const int tid = threadIdx.x, wave = tid >> 5, lane = tid & 31;
  const int hi8 = (lane >> 4) << 3, lo = lane & 15;
  const int nq0 = b * T + qt * 64;
  __shared__ __bf16 Qs[64 * 64];
  __shared__ __bf16 Ks[2][64 * 64];
  __shared__ __bf16 Vs[2][64 * 64];
  __shared__ __bf16 Ps[4 * 16 * 64];

  auto stage_kv = [&](int s) {
    const int nk0 = b * T + s * 64;
    const __bf16* kg = qkv + (size_t)nk0 * C3 + 768 + h * 64;
    const __bf16* vg = qkv + (size_t)nk0 * C3 + 1536 + h * 64;
#if HAVE_TDM
    if (tid < 32) {
      tdm_load_2d((unsigned)(size_t)&Ks[s & 1][0], kg, 64, 64, C3);
      tdm_load_2d((unsigned)(size_t)&Vs[s & 1][0], vg, 64, 64, C3);
    }
#else
    for (int t = tid; t < 1024; t += 128) {
      int row = t >> 4, g = t & 15;
      *(bf4*)&Ks[s & 1][row * 64 + g * 4] = *(const bf4*)&kg[(size_t)row * C3 + g * 4];
      *(bf4*)&Vs[s & 1][row * 64 + g * 4] = *(const bf4*)&vg[(size_t)row * C3 + g * 4];
    }
#endif
  };
  {
    const __bf16* qg = qkv + (size_t)nq0 * C3 + h * 64;
#if HAVE_TDM
    if (tid < 32) tdm_load_2d((unsigned)(size_t)&Qs[0], qg, 64, 64, C3);
#else
    for (int t = tid; t < 1024; t += 128) {
      int row = t >> 4, g = t & 15;
      *(bf4*)&Qs[row * 64 + g * 4] = *(const bf4*)&qg[(size_t)row * C3 + g * 4];
    }
#endif
  }
  stage_kv(0);
#if HAVE_TDM
  if (tid < 32) __builtin_amdgcn_s_wait_tensorcnt(0);
#endif
  __syncthreads();
  const v16bf qa0 = frag_ld(Qs, wave * 16 + lo, 64, 0, hi8);
  const v16bf qa1 = frag_ld(Qs, wave * 16 + lo, 64, 32, hi8);

  float m_r[8], l_r[8];
  v8f oacc[4] = {};
#pragma unroll
  for (int r = 0; r < 8; ++r) { m_r[r] = -__builtin_inff(); l_r[r] = 0.0f; }
  const float scale = 0.125f;  // 64^-0.5

  for (int kt = 0; kt < 16; ++kt) {
    if (kt + 1 < 16) stage_kv(kt + 1);
#if HAVE_TDM
    if (tid < 32) {
      if (kt + 1 < 16) __builtin_amdgcn_s_wait_tensorcnt(2);
      else             __builtin_amdgcn_s_wait_tensorcnt(0);
    }
#endif
    __syncthreads();
    const __bf16* Kb = Ks[kt & 1];
    const __bf16* Vb = Vs[kt & 1];

    v8f sacc[4];
#pragma unroll
    for (int nt = 0; nt < 4; ++nt) {                   // S = Q K^T (16x64)
      v16bf bk0 = frag_ld(Kb, nt * 16 + lo, 64, 0, hi8);
      v16bf bk1 = frag_ld(Kb, nt * 16 + lo, 64, 32, hi8);
      v8f s = {};
      s = wmma_bf16(qa0, bk0, s);
      s = wmma_bf16(qa1, bk1, s);
      sacc[nt] = s;
    }
    // Online softmax; C/D layout: lane holds column lo, rows hi8+r.
    float mnew[8], fct[8], rs[8];
#pragma unroll
    for (int r = 0; r < 8; ++r) {
      float t = sacc[0][r];
      t = fmaxf(t, sacc[1][r]); t = fmaxf(t, sacc[2][r]); t = fmaxf(t, sacc[3][r]);
      t *= scale;
#pragma unroll
      for (int msk = 1; msk < 16; msk <<= 1) t = fmaxf(t, __shfl_xor(t, msk, 32));
      mnew[r] = fmaxf(m_r[r], t);
      fct[r] = __expf(m_r[r] - mnew[r]);
      l_r[r] *= fct[r];
      rs[r] = 0.0f;
    }
#pragma unroll
    for (int dt = 0; dt < 4; ++dt)
#pragma unroll
      for (int r = 0; r < 8; ++r) oacc[dt][r] *= fct[r];

    __bf16* Pw = &Ps[wave * 1024];                     // per-wave P (16x64)
#pragma unroll
    for (int nt = 0; nt < 4; ++nt)
#pragma unroll
      for (int r = 0; r < 8; ++r) {
        float p = __expf(sacc[nt][r] * scale - mnew[r]);
        rs[r] += p;
        Pw[(hi8 + r) * 64 + nt * 16 + lo] = (__bf16)p;
      }
#pragma unroll
    for (int r = 0; r < 8; ++r) {
#pragma unroll
      for (int msk = 1; msk < 16; msk <<= 1) rs[r] += __shfl_xor(rs[r], msk, 32);
      l_r[r] += rs[r];
      m_r[r] = mnew[r];
    }
    v16bf pa0 = frag_ld(Pw, lo, 64, 0, hi8);           // P in A layout
    v16bf pa1 = frag_ld(Pw, lo, 64, 32, hi8);
#pragma unroll
    for (int dt = 0; dt < 4; ++dt) {                   // O += P V
      v16bf bv0 = frag_ld_col(Vb, dt * 16 + lo, 64, 0, hi8);
      v16bf bv1 = frag_ld_col(Vb, dt * 16 + lo, 64, 32, hi8);
      oacc[dt] = wmma_bf16(pa0, bv0, oacc[dt]);
      oacc[dt] = wmma_bf16(pa1, bv1, oacc[dt]);
    }
    __syncthreads();   // all reads of Ks/Vs[kt&1] done before re-targeting
  }
#pragma unroll
  for (int r = 0; r < 8; ++r) {
    float inv = 1.0f / l_r[r];
    int grow = nq0 + wave * 16 + hi8 + r;
#pragma unroll
    for (int dt = 0; dt < 4; ++dt)
      O[(size_t)grow * Cc + h * 64 + dt * 16 + lo] = (__bf16)(oacc[dt][r] * inv);
  }
}

// ---------------------------------------------------------------------------
extern "C" void kernel_launch(void* const* d_in, const int* in_sizes, int n_in,
                              void* d_out, int out_size, void* d_ws, size_t ws_size,
                              hipStream_t stream) {
  const int Mpad = 8, B = 8, T = 1024, C = 768, Hn = 12;
  const int Nrows = B * T;                 // 8192
  const float* feats = (const float*)d_in[0];
  const unsigned char* amask = (const unsigned char*)d_in[1];  // bool storage
  const float* Wqkv = (const float*)d_in[2];
  const float* Wp   = (const float*)d_in[3];
  const float* bp   = (const float*)d_in[4];
  float* out = (float*)d_out;

  // Workspace carve-up (bf16 buffers), ~67.6 MB total.
  char* ws = (char*)d_ws;
  __bf16* qkvbuf = (__bf16*)ws;                 ws += (size_t)Nrows * 3 * C * 2;
  __bf16* obuf   = (__bf16*)ws;                 ws += (size_t)Nrows * C * 2;
  __bf16* Xbf    = (__bf16*)ws;                 ws += (size_t)Nrows * C * 2;
  __bf16* Wqkvbf = (__bf16*)ws;                 ws += (size_t)3 * C * C * 2;
  __bf16* Wpbf   = (__bf16*)ws;

  // First M rows of output are zero.
  hipMemsetAsync(d_out, 0, (size_t)Mpad * C * sizeof(float), stream);

  // 0) f32 -> bf16 conversions.
  const int nX = Nrows * C, nWq = 3 * C * C, nWp = C * C;
  cvt_bf16_kernel<<<(nX / 4 + 255) / 256, 256, 0, stream>>>(
      feats + (size_t)Mpad * C, Xbf, nX / 4);
  cvt_bf16_kernel<<<(nWq / 4 + 255) / 256, 256, 0, stream>>>(Wqkv, Wqkvbf, nWq / 4);
  cvt_bf16_kernel<<<(nWp / 4 + 255) / 256, 256, 0, stream>>>(Wp, Wpbf, nWp / 4);

  // 1) QKV projection (masked), bf16 out.
  dim3 g1(Nrows / 128, (3 * C) / 128);
  qkv_gemm_kernel<<<g1, 256, 0, stream>>>(Xbf, Wqkvbf, amask, qkvbuf, C, 3 * C);
  // 2) Flash attention per (b, h, 64-query tile).
  dim3 g2(B * Hn, T / 64);
  attn_kernel<<<g2, 128, 0, stream>>>(qkvbuf, obuf);
  // 3) Output projection (+bias, masked), f32 out at row offset Mpad.
  dim3 g3(Nrows / 128, C / 128);
  proj_gemm_kernel<<<g3, 256, 0, stream>>>(obuf, Wpbf, bp, amask,
                                           out + (size_t)Mpad * C, C, C);
}
